// BEVEncoder_84645215470113
// MI455X (gfx1250) — matrix operand order repaired and checked by reading, not compiled, not run on hardware
//
#include <hip/hip_runtime.h>
#include <hip/hip_bf16.h>
#include <hip/hip_fp16.h>

typedef __attribute__((ext_vector_type(16))) _Float16 v16h;
typedef __attribute__((ext_vector_type(8)))  _Float16 v8h;
typedef __attribute__((ext_vector_type(8)))  float    v8f;

// ---------------------------------------------------------------- utilities

__global__ void zero_f32_kernel(float* __restrict__ p, int n) {
  int i = blockIdx.x * blockDim.x + threadIdx.x;
  if (i < n) p[i] = 0.0f;
}

// ------------------------------------------------ weight repack (f32 -> f16)
// Packs conv weights [Cout][CIN][ks][ks] into WMMA A-fragment order:
//   packed[(((mt*K2 + tap)*(CIN/32) + cb)*32 + lane)*16 + i]
// so each lane's 16-half A fragment is one contiguous aligned 32B load and a
// wave reads 1KB contiguously per chunk. Fragment layout (contiguous k order, not the ISA table order):
//   row = mt*16 + (lane&15); k = (lane>>4)*16 + i.
__global__ void repack_w_kernel(const float* __restrict__ w, _Float16* __restrict__ out,
                                int Cout, int CIN, int ksize) {
  const int K2 = ksize * ksize;
  const int total = Cout * CIN * K2;
  int idx = blockIdx.x * blockDim.x + threadIdx.x;
  if (idx >= total) return;
  int i = idx & 15;
  int lane = (idx >> 4) & 31;
  int chunk = idx >> 9;
  int nchunks_c = CIN >> 5;
  int cb = chunk % nchunks_c;
  int mt_tap = chunk / nchunks_c;
  int tap = mt_tap % K2;
  int mt = mt_tap / K2;
  int kh = tap / ksize, kw = tap % ksize;
  int co = mt * 16 + (lane & 15);
  int kb = (lane >> 4) << 4;
  int klocal = kb + i;
  int ci = (cb << 5) + klocal;
  out[idx] = (_Float16)w[((co * CIN + ci) * ksize + kh) * ksize + kw];
}

// --------------------------------------------------- lidar point scatter-max
// bev layout: [5][200][200]; ch 0..3 = height bins (value z+2), 4 = intensity.
// Grid pre-zeroed; only positive values can win vs 0, so int-punned atomicMax
// (order-preserving for non-negative floats) matches jnp .at[].max.
__global__ void scatter_kernel(const float* __restrict__ pts, int n, float* __restrict__ bev) {
  int i = blockIdx.x * blockDim.x + threadIdx.x;
  if (i >= n) return;
  float x = pts[i * 4 + 0], y = pts[i * 4 + 1], z = pts[i * 4 + 2], it = pts[i * 4 + 3];
  int xi = (int)floorf((x + 50.0f) * 2.0f);
  int yi = (int)floorf((y + 50.0f) * 2.0f);
  xi = min(max(xi, 0), 199);
  yi = min(max(yi, 0), 199);
  int hi = (z > -2.0f) + (z > 0.0f) + (z > 2.0f);   // searchsorted(left) clipped to 3
  float v = z + 2.0f;
  if (v > 0.0f)
    atomicMax((int*)(bev + (size_t)hi * 40000 + yi * 200 + xi), __float_as_int(v));
  if (it > 0.0f)
    atomicMax((int*)(bev + (size_t)4 * 40000 + yi * 200 + xi), __float_as_int(it));
}

// ------------------------------------------- direct conv (Cin=3/5, Cout=32)
// Thread-per-pixel, 32 output channels in registers, weights in LDS.
// Input NCHW f32, output NHWC f16 (contiguous 32-half store per thread).
__global__ __launch_bounds__(256) void conv3x3_direct_kernel(
    const float* __restrict__ X, const float* __restrict__ Wt,
    const float* __restrict__ bias, const float* __restrict__ g,
    const float* __restrict__ beta, const float* __restrict__ mean,
    const float* __restrict__ var, _Float16* __restrict__ out,
    int CIN, int H, int W, int OH, int OW, int stride) {
  extern __shared__ float sw[];
  const int nw = 32 * CIN * 9;
  for (int i = threadIdx.x; i < nw; i += blockDim.x) sw[i] = Wt[i];
  __syncthreads();
  int px = blockIdx.x * blockDim.x + threadIdx.x;
  int npix = OH * OW;
  if (px >= npix) return;
  int oy = px / OW, ox = px % OW;
  float acc[32];
#pragma unroll
  for (int c = 0; c < 32; ++c) acc[c] = 0.0f;
  for (int ci = 0; ci < CIN; ++ci) {
    for (int kh = 0; kh < 3; ++kh) {
      int iy = oy * stride - 1 + kh;
      if (iy < 0 || iy >= H) continue;
      for (int kw = 0; kw < 3; ++kw) {
        int ix = ox * stride - 1 + kw;
        if (ix < 0 || ix >= W) continue;
        float xv = X[(size_t)ci * H * W + (size_t)iy * W + ix];
        const float* wp = sw + (ci * 3 + kh) * 3 + kw;
#pragma unroll
        for (int c = 0; c < 32; ++c) acc[c] += xv * wp[c * CIN * 9];
      }
    }
  }
  _Float16* op = out + (size_t)px * 32;
#pragma unroll
  for (int c = 0; c < 32; ++c) {
    float s = g[c] * rsqrtf(var[c] + 1e-5f);
    float yv = fmaxf(acc[c] * s + (bias[c] * s + beta[c] - mean[c] * s), 0.0f);
    op[c] = (_Float16)yv;
  }
}

// ------------------------------------- implicit-GEMM conv via WMMA f16->f32
// GEMM: M = Cout, N = output pixels, K = Cin*ks*ks (Cin % 32 == 0).
// Activations NHWC f16; weights pre-packed into A-fragment order.
// Each wave: 16-pixel tile x 64 output channels (4 accumulators, B reused 4x).
// B fragment = one contiguous aligned v16h load per lane (lane&15 = pixel col,
// (lane>>4)*16 = channel half). D stores: 8 contiguous channels per acc row.
// out_layout: 0 = NHWC f16 (out_h), 1 = NHWC f32 (out_f), 2 = NCHW f32 (out_f).
__global__ __launch_bounds__(256) void conv_wmma_kernel(
    const _Float16* __restrict__ X, const _Float16* __restrict__ Wt,
    const float* __restrict__ bias, const float* __restrict__ g,
    const float* __restrict__ beta, const float* __restrict__ mean,
    const float* __restrict__ var,
    _Float16* __restrict__ out_h, float* __restrict__ out_f,
    int CIN, int H, int W, int OH, int OW, int Cout,
    int ksize, int stride, int pad, int bn_relu, int out_layout) {
  const int lane = threadIdx.x & 31;
  const int wid = __builtin_amdgcn_readfirstlane((blockIdx.x * blockDim.x + threadIdx.x) >> 5);
  const int npix = OH * OW;
  const int ntiles = npix >> 4;
  const int mgroups = Cout >> 6;
  if (wid >= ntiles * mgroups) return;   // scalar-uniform exit: EXEC all-ones below
  const int ntile = wid % ntiles;
  const int mg = wid / ntiles;

  const int px = (ntile << 4) + (lane & 15);
  const int oy = px / OW;
  const int ox = px % OW;
  const int kbb = (lane >> 4) << 4;
  const int K2 = ksize * ksize;
  const int nchunks_c = CIN >> 5;

  v8f acc[4];
#pragma unroll
  for (int mi = 0; mi < 4; ++mi)
#pragma unroll
    for (int r = 0; r < 8; ++r) acc[mi][r] = 0.0f;

  for (int tap = 0; tap < K2; ++tap) {
    const int kh = tap / ksize, kw = tap % ksize;
    const int iy = oy * stride - pad + kh;
    const int ix = ox * stride - pad + kw;
    const bool inb = (iy >= 0) && (iy < H) && (ix >= 0) && (ix < W);
    const _Float16* brow = X + (size_t)(iy * W + ix) * CIN + kbb;
    if (tap + 1 < K2) {  // prefetch next tap's B row (global_prefetch_b8)
      const int kh2 = (tap + 1) / ksize, kw2 = (tap + 1) % ksize;
      const int iy2 = oy * stride - pad + kh2;
      const int ix2 = ox * stride - pad + kw2;
      if (iy2 >= 0 && iy2 < H && ix2 >= 0 && ix2 < W)
        __builtin_prefetch(X + (size_t)(iy2 * W + ix2) * CIN + kbb, 0, 3);
    }
    for (int cib = 0; cib < CIN; cib += 32) {
      v16h bfrag;
      if (inb) {
        bfrag = *(const v16h*)(brow + cib);           // 32B aligned contiguous
      } else {
#pragma unroll
        for (int i = 0; i < 16; ++i) bfrag[i] = (_Float16)0.0f;
      }
      const size_t chunk_base =
          ((size_t)((mg << 2) * K2 + tap) * nchunks_c + (cib >> 5)) << 9;
#pragma unroll
      for (int mi = 0; mi < 4; ++mi) {
        const size_t coff = chunk_base + ((size_t)mi * K2 * nchunks_c << 9);
        v16h afrag = *(const v16h*)(Wt + coff + ((size_t)lane << 4));
        acc[mi] = __builtin_amdgcn_wmma_f32_16x16x32_f16(
            false, afrag, false, bfrag, (short)0, acc[mi], false, false);
      }
    }
  }

  const int rowoff = (lane >> 4) << 3;
#pragma unroll
  for (int mi = 0; mi < 4; ++mi) {
    const int cobase = ((mg << 2) + mi) * 16 + rowoff;
    float vals[8];
#pragma unroll
    for (int r = 0; r < 8; ++r) {
      const int co = cobase + r;
      float a = acc[mi][r];
      if (bn_relu) {
        float s = g[co] * rsqrtf(var[co] + 1e-5f);
        vals[r] = fmaxf(a * s + (bias[co] * s + beta[co] - mean[co] * s), 0.0f);
      } else {
        vals[r] = a + bias[co];
      }
    }
    if (out_layout == 0) {
      v8h hv;
#pragma unroll
      for (int r = 0; r < 8; ++r) hv[r] = (_Float16)vals[r];
      *(v8h*)(out_h + (size_t)px * Cout + cobase) = hv;        // 16B aligned
    } else if (out_layout == 1) {
      v8f fv;
#pragma unroll
      for (int r = 0; r < 8; ++r) fv[r] = vals[r];
      *(v8f*)(out_f + (size_t)px * Cout + cobase) = fv;        // 32B aligned
    } else {
#pragma unroll
      for (int r = 0; r < 8; ++r)
        out_f[(size_t)(cobase + r) * npix + px] = vals[r];     // NCHW final
    }
  }
}

// ---------------------- bilinear resize (JAX half-pixel), NHWC in -> NCHW out
// Thread handles 8 channels of one output pixel: 4x contiguous 8-float loads.
__global__ void resize_kernel(const float* __restrict__ in, float* __restrict__ out,
                              int C, int IH, int IW, int OH, int OW) {
  int idx = blockIdx.x * blockDim.x + threadIdx.x;
  int nc8 = C >> 3;
  int total = OH * OW * nc8;
  if (idx >= total) return;
  int c8 = idx & (nc8 - 1);
  int px = idx / nc8;
  int oy = px / OW, ox = px % OW;
  float fy = (oy + 0.5f) * ((float)IH / (float)OH) - 0.5f;
  float fx = (ox + 0.5f) * ((float)IW / (float)OW) - 0.5f;
  int y0 = (int)floorf(fy), x0 = (int)floorf(fx);
  float wy = fy - (float)y0, wx = fx - (float)x0;
  int y0c = min(max(y0, 0), IH - 1), y1c = min(max(y0 + 1, 0), IH - 1);
  int x0c = min(max(x0, 0), IW - 1), x1c = min(max(x0 + 1, 0), IW - 1);
  const float* p00 = in + ((size_t)y0c * IW + x0c) * C + c8 * 8;
  const float* p01 = in + ((size_t)y0c * IW + x1c) * C + c8 * 8;
  const float* p10 = in + ((size_t)y1c * IW + x0c) * C + c8 * 8;
  const float* p11 = in + ((size_t)y1c * IW + x1c) * C + c8 * 8;
  v8f a = *(const v8f*)p00, b = *(const v8f*)p01;
  v8f c = *(const v8f*)p10, d = *(const v8f*)p11;
#pragma unroll
  for (int r = 0; r < 8; ++r) {
    float v = (1.0f - wy) * ((1.0f - wx) * a[r] + wx * b[r]) +
              wy          * ((1.0f - wx) * c[r] + wx * d[r]);
    out[(size_t)(c8 * 8 + r) * (OH * OW) + px] = v;
  }
}

// ----------------------------------------------------------------- launcher

extern "C" void kernel_launch(void* const* d_in, const int* in_sizes, int n_in,
                              void* d_out, int out_size, void* d_ws, size_t ws_size,
                              hipStream_t stream) {
  (void)in_sizes; (void)n_in; (void)out_size; (void)ws_size;
  const float* images = (const float*)d_in[0];
  const float* points = (const float*)d_in[1];
  const float* c1w  = (const float*)d_in[2];  const float* c1b  = (const float*)d_in[3];
  const float* bn1g = (const float*)d_in[4];  const float* bn1b = (const float*)d_in[5];
  const float* bn1m = (const float*)d_in[6];  const float* bn1v = (const float*)d_in[7];
  const float* c2w  = (const float*)d_in[8];  const float* c2b  = (const float*)d_in[9];
  const float* bn2g = (const float*)d_in[10]; const float* bn2b = (const float*)d_in[11];
  const float* bn2m = (const float*)d_in[12]; const float* bn2v = (const float*)d_in[13];
  const float* c3w  = (const float*)d_in[14]; const float* c3b  = (const float*)d_in[15];
  const float* bn3g = (const float*)d_in[16]; const float* bn3b = (const float*)d_in[17];
  const float* bn3m = (const float*)d_in[18]; const float* bn3v = (const float*)d_in[19];
  const float* p1w  = (const float*)d_in[20]; const float* p1b  = (const float*)d_in[21];
  const float* pbg  = (const float*)d_in[22]; const float* pbb  = (const float*)d_in[23];
  const float* pbm  = (const float*)d_in[24]; const float* pbv  = (const float*)d_in[25];
  const float* p2w  = (const float*)d_in[26]; const float* p2b  = (const float*)d_in[27];
  const float* lc1w = (const float*)d_in[28]; const float* lc1b = (const float*)d_in[29];
  const float* lb1g = (const float*)d_in[30]; const float* lb1b = (const float*)d_in[31];
  const float* lb1m = (const float*)d_in[32]; const float* lb1v = (const float*)d_in[33];
  const float* lc2w = (const float*)d_in[34]; const float* lc2b = (const float*)d_in[35];
  const float* lb2g = (const float*)d_in[36]; const float* lb2b = (const float*)d_in[37];
  const float* lb2m = (const float*)d_in[38]; const float* lb2v = (const float*)d_in[39];
  const float* lc3w = (const float*)d_in[40]; const float* lc3b = (const float*)d_in[41];

  char* ws = (char*)d_ws;
  size_t cur = 0;
  auto alloc = [&](size_t bytes) { size_t o = cur; cur += (bytes + 255) & ~(size_t)255; return o; };

  _Float16* wpool = (_Float16*)(ws + alloc(446464ull * 2));
  _Float16* w_c2  = wpool;            // 18432
  _Float16* w_c3  = wpool + 18432;    // 73728
  _Float16* w_p1  = wpool + 92160;    // 294912
  _Float16* w_p2  = wpool + 387072;   // 32768
  _Float16* w_lc2 = wpool + 419840;   // 18432
  _Float16* w_lc3 = wpool + 438272;   // 8192
  float*    bev   = (float*)(ws + alloc(200000ull * 4));
  _Float16* lf1   = (_Float16*)(ws + alloc(32ull * 40000 * 2));   // NHWC
  _Float16* lf2   = (_Float16*)(ws + alloc(64ull * 40000 * 2));   // NHWC
  char*     regA  = ws + alloc(33554432ull);  // f1 (33.5MB) -> then f3 + p2out
  char*     regB  = ws + alloc(16777216ull);  // f2 (16.8MB) -> then p1out
  _Float16* camf1 = (_Float16*)regA;                 // NHWC 512x1024x32
  _Float16* camf3 = (_Float16*)regA;                 // aliases f1 (dead after c2)
  float*    p2o   = (float*)(regA + 8388608);        // NHWC 128x256x128 f32
  _Float16* camf2 = (_Float16*)regB;                 // NHWC 256x512x64
  _Float16* p1o   = (_Float16*)regB;                 // aliases f2 (dead after c3)

  float* out_cam = (float*)d_out;
  float* out_lid = (float*)d_out + (size_t)128 * 40000;

  // ---- lidar BEV rasterization
  zero_f32_kernel<<<(200000 + 255) / 256, 256, 0, stream>>>(bev, 200000);
  scatter_kernel<<<(20000 + 255) / 256, 256, 0, stream>>>(points, 20000, bev);

  // ---- weight repack into WMMA A-fragment order (f32 -> f16)
  repack_w_kernel<<<(18432 + 255) / 256, 256, 0, stream>>>(c2w, w_c2, 64, 32, 3);
  repack_w_kernel<<<(73728 + 255) / 256, 256, 0, stream>>>(c3w, w_c3, 128, 64, 3);
  repack_w_kernel<<<(294912 + 255) / 256, 256, 0, stream>>>(p1w, w_p1, 256, 128, 3);
  repack_w_kernel<<<(32768 + 255) / 256, 256, 0, stream>>>(p2w, w_p2, 128, 256, 1);
  repack_w_kernel<<<(18432 + 255) / 256, 256, 0, stream>>>(lc2w, w_lc2, 64, 32, 3);
  repack_w_kernel<<<(8192 + 255) / 256, 256, 0, stream>>>(lc3w, w_lc3, 128, 64, 1);

  // ---- first layers (tiny Cin): direct fp32 conv, NHWC f16 out
  conv3x3_direct_kernel<<<(524288 + 255) / 256, 256, 3 * 32 * 9 * 4, stream>>>(
      images, c1w, c1b, bn1g, bn1b, bn1m, bn1v, camf1, 3, 1024, 2048, 512, 1024, 2);
  conv3x3_direct_kernel<<<(40000 + 255) / 256, 256, 5 * 32 * 9 * 4, stream>>>(
      bev, lc1w, lc1b, lb1g, lb1b, lb1m, lb1v, lf1, 5, 200, 200, 200, 200, 1);

  auto wmma_launch = [&](const _Float16* X, const _Float16* Wt, const float* b,
                         const float* gg, const float* bb, const float* mm, const float* vv,
                         _Float16* oh, float* of, int CIN, int H, int W, int OH, int OW,
                         int Cout, int ks, int st, int pd, int rb, int layout) {
    long waves = (long)(Cout / 64) * (long)((OH * OW) / 16);
    int blocks = (int)((waves * 32 + 255) / 256);
    conv_wmma_kernel<<<blocks, 256, 0, stream>>>(X, Wt, b, gg, bb, mm, vv, oh, of,
                                                 CIN, H, W, OH, OW, Cout, ks, st, pd, rb, layout);
  };

  // ---- camera WMMA layers
  wmma_launch(camf1, w_c2, c2b, bn2g, bn2b, bn2m, bn2v, camf2, nullptr,
              32, 512, 1024, 256, 512, 64, 3, 2, 1, 1, 0);
  wmma_launch(camf2, w_c3, c3b, bn3g, bn3b, bn3m, bn3v, camf3, nullptr,
              64, 256, 512, 128, 256, 128, 3, 2, 1, 1, 0);
  wmma_launch(camf3, w_p1, p1b, pbg, pbb, pbm, pbv, p1o, nullptr,
              128, 128, 256, 128, 256, 256, 3, 1, 1, 1, 0);
  wmma_launch(p1o, w_p2, p2b, nullptr, nullptr, nullptr, nullptr, nullptr, p2o,
              256, 128, 256, 128, 256, 128, 1, 1, 0, 0, 1);

  // ---- lidar WMMA layers
  wmma_launch(lf1, w_lc2, lc2b, lb2g, lb2b, lb2m, lb2v, lf2, nullptr,
              32, 200, 200, 200, 200, 64, 3, 1, 1, 1, 0);
  wmma_launch(lf2, w_lc3, lc3b, nullptr, nullptr, nullptr, nullptr, nullptr, out_lid,
              64, 200, 200, 200, 200, 128, 1, 1, 0, 0, 2);

  // ---- camera branch resize into output (NHWC -> NCHW)
  resize_kernel<<<(40000 * 16 + 255) / 256, 256, 0, stream>>>(
      p2o, out_cam, 128, 128, 256, 200, 200);
}
